// S4Model_36172214567047
// MI455X (gfx1250) — compile-verified, matrix operand order checked by reading, not run on hardware
//
#include <hip/hip_runtime.h>
#include <math.h>

#define Bc 8
#define Lc 4096
#define Hc 512
#define Nc 16
#define NLc 4
#define Mc (Bc*Lc)

typedef __attribute__((ext_vector_type(16))) __bf16 v16bf;
typedef __attribute__((ext_vector_type(8)))  float  v8f;
typedef __attribute__((ext_vector_type(8)))  __bf16 bf16x8;
typedef __attribute__((ext_vector_type(4)))  unsigned int v4u;
typedef __attribute__((ext_vector_type(8)))  int v8i;
typedef __attribute__((ext_vector_type(4)))  int v4i;

union FragBF { v16bf v; bf16x8 h[2]; };

// ---------------- parameter precompute: a = exp(dt*A), Ct = C*(exp(dt*A)-1)/A --------------
__global__ void __launch_bounds__(256) precompute_params(
    const float* __restrict__ log_dt, const float* __restrict__ Are,
    const float* __restrict__ Aim, const float* __restrict__ Cre,
    const float* __restrict__ Cim, float* __restrict__ P) {
  int idx = blockIdx.x * 256 + threadIdx.x;          // [0, NL*H*N)
  int n  = idx % Nc;
  int t  = idx / Nc;
  int h  = t % Hc;
  int il = t / Hc;
  float dt = __expf(log_dt[il * Hc + h]);
  float Ar = Are[idx], Ai = Aim[idx];
  float dr = Ar * dt, di = Ai * dt;
  float mg = __expf(dr);
  float er = mg * __cosf(di), ei = mg * __sinf(di);  // a = exp(dt*A)
  float nr = er - 1.f, ni = ei;                      // exp(dt*A) - 1
  float den = Ar * Ar + Ai * Ai;
  float wr = (nr * Ar + ni * Ai) / den;              // (e-1)/A
  float wi = (ni * Ar - nr * Ai) / den;
  const int T = NLc * Hc * Nc;
  P[idx] = er;
  P[T + idx] = ei;
  for (int c = 0; c < 2; ++c) {
    int ci = ((il * 2 + c) * Hc + h) * Nc + n;
    float Cr = Cre[ci], Ci = Cim[ci];
    P[(2 + 2 * c) * T + idx] = Cr * wr - Ci * wi;    // Ct real
    P[(3 + 2 * c) * T + idx] = Cr * wi + Ci * wr;    // Ct imag
  }
}

// ---------------- weights fp32 -> bf16 --------------------------------------------------
__global__ void __launch_bounds__(256) to_bf16(const float* __restrict__ w,
                                               __bf16* __restrict__ o, int n) {
  int i = blockIdx.x * 256 + threadIdx.x;
  if (i < n) o[i] = (__bf16)w[i];
}

// ---------------- encoder: h = [x, grid] @ enc_w^T + enc_b ------------------------------
__global__ void __launch_bounds__(256) encode(const float* __restrict__ x,
                                              const float* __restrict__ ew,
                                              const float* __restrict__ eb,
                                              float* __restrict__ h) {
  size_t idx = (size_t)blockIdx.x * 256 + threadIdx.x;  // [0, M*H)
  int col = (int)(idx & (Hc - 1));
  size_t row = idx >> 9;                                // b*L + l
  int l = (int)(row & (Lc - 1));
  float xv = x[row];
  float g = (float)l * (1.f / (float)(Lc - 1));
  h[idx] = ew[col * 2] * xv + ew[col * 2 + 1] * g + eb[col];
}

// ---------------- layernorm over H=512, one row per block -------------------------------
__global__ void __launch_bounds__(256) layernorm(const float* __restrict__ h,
                                                 const float* __restrict__ w,
                                                 const float* __restrict__ b,
                                                 float* __restrict__ z) {
  __shared__ float s1[256], s2[256];
  int tid = threadIdx.x;
  size_t base = (size_t)blockIdx.x * Hc;
  float a0 = h[base + tid], a1 = h[base + tid + 256];
  s1[tid] = a0 + a1;
  s2[tid] = a0 * a0 + a1 * a1;
  __syncthreads();
  for (int off = 128; off > 0; off >>= 1) {
    if (tid < off) { s1[tid] += s1[tid + off]; s2[tid] += s2[tid + off]; }
    __syncthreads();
  }
  float mu = s1[0] * (1.f / Hc);
  float var = s2[0] * (1.f / Hc) - mu * mu;
  float inv = rsqrtf(var + 1e-5f);
  z[base + tid]       = (a0 - mu) * inv * w[tid]       + b[tid];
  z[base + tid + 256] = (a1 - mu) * inv * w[tid + 256] + b[tid + 256];
}

// ---------------- transpose (B,L,H) f32 -> (B,H,L) f32 ----------------------------------
__global__ void __launch_bounds__(256) transpose_f32(const float* __restrict__ in,
                                                     float* __restrict__ out) {
  __shared__ float tile[32][33];
  int b = blockIdx.z;
  int h0 = blockIdx.x * 32, l0 = blockIdx.y * 32;
  int tx = threadIdx.x, ty = threadIdx.y;
  for (int j = 0; j < 32; j += 8)
    tile[ty + j][tx] = in[((size_t)b * Lc + (l0 + ty + j)) * Hc + h0 + tx];
  __syncthreads();
  for (int j = 0; j < 32; j += 8)
    out[((size_t)b * Hc + (h0 + ty + j)) * Lc + l0 + tx] = tile[tx][ty + j];
}

// ---------------- transpose (B,H,L) bf16 -> (B,L,H) bf16 --------------------------------
__global__ void __launch_bounds__(256) transpose_bf16(const __bf16* __restrict__ in,
                                                      __bf16* __restrict__ out) {
  __shared__ __bf16 tile[32][34];
  int b = blockIdx.z;
  int l0 = blockIdx.x * 32, h0 = blockIdx.y * 32;
  int tx = threadIdx.x, ty = threadIdx.y;
  for (int j = 0; j < 32; j += 8)
    tile[ty + j][tx] = in[((size_t)b * Hc + (h0 + ty + j)) * Lc + l0 + tx];
  __syncthreads();
  for (int j = 0; j < 32; j += 8)
    out[((size_t)b * Lc + (l0 + ty + j)) * Hc + h0 + tx] = tile[tx][ty + j];
}

// ---------------- bidirectional diagonal SSM scan + D*u skip + exact GELU ---------------
// One wave serves 2 channels (lanes 0-15 / 16-31), 16 lanes = 16 complex states.
// Block: 64 threads (2 waves) = 4 channels; full rows staged in 128KB dynamic LDS.
// Output is bf16 (the GEMM A-matrix precision) to halve store/transpose bandwidth.
__global__ void __launch_bounds__(64) s4_scan(
    const float* __restrict__ zT,      // (B*H, L)
    const float* __restrict__ are, const float* __restrict__ aim,
    const float* __restrict__ c0r, const float* __restrict__ c0i,
    const float* __restrict__ c1r, const float* __restrict__ c1i,
    const float* __restrict__ Dp,      // (H,)
    __bf16* __restrict__ yT) {         // (B*H, L) bf16
  extern __shared__ float lds[];
  float* u = lds;              // [4*L]
  float* y = lds + 4 * Lc;     // [4*L]
  int tid = threadIdx.x;
  size_t chBase = (size_t)blockIdx.x * 4;
  for (int idx = tid; idx < 4 * Lc; idx += 64)
    u[idx] = zT[chBase * Lc + idx];
  __syncthreads();

  int lane = tid & 31;
  int wave = tid >> 5;
  int cl = wave * 2 + (lane >> 4);             // local channel 0..3
  int hh = (int)((chBase + cl) & (Hc - 1));    // channel's h index
  int n = lane & 15;
  int pi = hh * Nc + n;
  float ar = are[pi], ai = aim[pi];
  float C0r = c0r[pi], C0i = c0i[pi];
  float C1r = c1r[pi], C1i = c1i[pi];
  const float* uc = u + cl * Lc;
  float* yc = y + cl * Lc;

  // forward causal scan: y[t] = 2*Re(C0 . x[t]), x[t] = a*x[t-1] + u[t]
  float sr = 0.f, si = 0.f;
  for (int t = 0; t < Lc; ++t) {
    float uv = uc[t];
    float nr = ar * sr - ai * si + uv;
    float ni = ar * si + ai * sr;
    sr = nr; si = ni;
    float p = C0r * sr - C0i * si;
    p += __shfl_xor(p, 1, 16);
    p += __shfl_xor(p, 2, 16);
    p += __shfl_xor(p, 4, 16);
    p += __shfl_xor(p, 8, 16);
    if (n == 0) yc[t] = 2.f * p;
  }
  // anticausal scan: y[t-1] += 2*Re(C1 . xb), xb = a*xb + u[t], t descending
  sr = 0.f; si = 0.f;
  for (int t = Lc - 1; t >= 1; --t) {
    float uv = uc[t];
    float nr = ar * sr - ai * si + uv;
    float ni = ar * si + ai * sr;
    sr = nr; si = ni;
    float p = C1r * sr - C1i * si;
    p += __shfl_xor(p, 1, 16);
    p += __shfl_xor(p, 2, 16);
    p += __shfl_xor(p, 4, 16);
    p += __shfl_xor(p, 8, 16);
    if (n == 0) yc[t - 1] += 2.f * p;
  }
  __syncthreads();

  // epilogue: skip connection + exact GELU, coalesced bf16 store
  for (int idx = tid; idx < 4 * Lc; idx += 64) {
    int hcur = (int)((chBase + (idx >> 12)) & (Hc - 1));
    float v = y[idx] + Dp[hcur] * u[idx];
    v = 0.5f * v * (1.f + erff(v * 0.70710678118654752f));
    yT[chBase * Lc + idx] = (__bf16)v;
  }
}

// ---------------- fused output GEMM (bf16 WMMA) + bias + GLU + residual -----------------
// Value+gate weight panels (16 rows x 512 each) are DMAed into LDS once per workgroup by
// the Tensor Data Mover (tensor_load_to_lds, TENSORcnt), then all 8 waves read B
// fragments from LDS; A fragments stream from global (L2-resident).
__global__ void __launch_bounds__(256) gemm_glu(
    const __bf16* __restrict__ A,    // (M, 512) bf16 activations
    const __bf16* __restrict__ W,    // (1024, 512) bf16 weights, row-major (out, in)
    const float* __restrict__ bias,  // (1024,)
    float* __restrict__ h) {         // (M, 512) residual stream, updated in place
  __shared__ __align__(16) __bf16 sB[2 * 16 * Hc];   // 32KB: value rows then gate rows
  const int lane = threadIdx.x & 31;
  const int wave = threadIdx.x >> 5;
  const int m0 = blockIdx.x * 128 + wave * 16;
  const int n0 = blockIdx.y * 16;
  const int nl = lane & 15;
  const int hi = lane >> 4;

  if (threadIdx.x == 0) {
    // D# group1: workgroup_mask=0, data_size=1 (2B), tensor 512x16, tile 512x16, stride 512
    v8i g1 = { (int)(1u << 16),        // data_size = 2 bytes
               (int)(512u << 16),      // tensor_dim0 lo16 in [63:48]
               (int)(16u << 16),       // tensor_dim0 hi=0 | tensor_dim1=16
               (int)(512u << 16),      // tensor_dim1 hi=0 | tile_dim0=512
               16,                     // tile_dim1=16 | tile_dim2=0
               512,                    // tensor_dim0_stride lo32
               0, 0 };                 // stride hi / dim1_stride
    v4i gz4 = { 0, 0, 0, 0 };
    v8i gz8 = { 0, 0, 0, 0, 0, 0, 0, 0 };
    unsigned lds0 = (unsigned)(size_t)(void*)&sB[0];
    unsigned lds1 = (unsigned)(size_t)(void*)&sB[16 * Hc];
    unsigned long long ga0 = (unsigned long long)(size_t)(W + (size_t)n0 * Hc);
    unsigned long long ga1 = (unsigned long long)(size_t)(W + (size_t)(Hc + n0) * Hc);
    v4u g0a = { 1u, lds0, (unsigned)ga0, (unsigned)(ga0 >> 32) | (2u << 30) };
    v4u g0b = { 1u, lds1, (unsigned)ga1, (unsigned)(ga1 >> 32) | (2u << 30) };
    __builtin_amdgcn_tensor_load_to_lds(g0a, g1, gz4, gz4, gz8, 0);
    __builtin_amdgcn_tensor_load_to_lds(g0b, g1, gz4, gz4, gz8, 0);
    __builtin_amdgcn_s_wait_tensorcnt(0);
  }
  __syncthreads();

  v8f cv = {};
  v8f cg = {};
  const __bf16* arow  = A + (size_t)(m0 + nl) * Hc;
  const __bf16* wvrow = sB + (size_t)nl * Hc;
  const __bf16* wgrow = sB + (size_t)(16 + nl) * Hc;

  for (int k = 0; k < Hc; k += 32) {
    const int o0 = k + hi * 8;
    const int o1 = k + 16 + hi * 8;
    FragBF a, bv, bg;
    a.h[0]  = *(const bf16x8*)(arow  + o0);  a.h[1]  = *(const bf16x8*)(arow  + o1);
    bv.h[0] = *(const bf16x8*)(wvrow + o0);  bv.h[1] = *(const bf16x8*)(wvrow + o1);
    bg.h[0] = *(const bf16x8*)(wgrow + o0);  bg.h[1] = *(const bf16x8*)(wgrow + o1);
    if (k + 32 < Hc) __builtin_prefetch(arow + o0 + 32, 0, 0);
    cv = __builtin_amdgcn_wmma_f32_16x16x32_bf16(false, a.v, false, bv.v, (short)0, cv, false, false);
    cg = __builtin_amdgcn_wmma_f32_16x16x32_bf16(false, a.v, false, bg.v, (short)0, cg, false, false);
  }

  const float bvs = bias[n0 + nl];
  const float bgs = bias[Hc + n0 + nl];
  for (int r = 0; r < 8; ++r) {
    int m = m0 + r + 8 * hi;            // C/D layout: lane = N, vgpr r = M (+8 for hi lanes)
    size_t idx = (size_t)m * Hc + n0 + nl;
    float val  = cv[r] + bvs;
    float gate = cg[r] + bgs;
    float res = val * (1.f / (1.f + __expf(-gate)));   // GLU
    h[idx] = h[idx] + res;                             // residual
  }
}

// ---------------- decoder: out[b,l] = h[b,l,:] . dec_w + dec_b --------------------------
__global__ void __launch_bounds__(256) decode(const float* __restrict__ h,
                                              const float* __restrict__ dw,
                                              const float* __restrict__ db,
                                              float* __restrict__ out) {
  int lane = threadIdx.x & 31;
  int row = blockIdx.x * 8 + (threadIdx.x >> 5);
  const float* x = h + (size_t)row * Hc;
  float acc = 0.f;
  for (int k = lane; k < Hc; k += 32) acc += x[k] * dw[k];
  for (int m = 16; m > 0; m >>= 1) acc += __shfl_xor(acc, m, 32);
  if (lane == 0) out[row] = acc + db[0];
}

extern "C" void kernel_launch(void* const* d_in, const int* in_sizes, int n_in,
                              void* d_out, int out_size, void* d_ws, size_t ws_size,
                              hipStream_t stream) {
  (void)in_sizes; (void)n_in; (void)out_size; (void)ws_size;
  const float* x     = (const float*)d_in[0];
  const float* enc_w = (const float*)d_in[1];
  const float* enc_b = (const float*)d_in[2];
  const float* logdt = (const float*)d_in[3];
  const float* Are   = (const float*)d_in[4];
  const float* Aim   = (const float*)d_in[5];
  const float* Cre   = (const float*)d_in[6];
  const float* Cim   = (const float*)d_in[7];
  const float* Dp    = (const float*)d_in[8];
  const float* outw  = (const float*)d_in[9];
  const float* outb  = (const float*)d_in[10];
  const float* lnw   = (const float*)d_in[11];
  const float* lnb   = (const float*)d_in[12];
  const float* decw  = (const float*)d_in[13];
  const float* decb  = (const float*)d_in[14];
  float* out = (float*)d_out;

  char* ws = (char*)d_ws;
  const size_t EL = (size_t)Mc * Hc;                 // 16,777,216 elements
  float*  hbuf = (float*)(ws);                       // residual stream (B,L,H)
  float*  bufA = (float*)(ws + EL * 4);              // z (B,L,H) f32 / yT (B,H,L) bf16
  float*  bufB = (float*)(ws + EL * 8);              // zT (B,H,L) f32
  __bf16* bufC = (__bf16*)(ws + EL * 12);            // GEMM A (B,L,H) bf16
  float*  P    = (float*)(ws + EL * 12 + EL * 2);    // SSM params: 6 arrays of NL*H*N
  __bf16* wb   = (__bf16*)(ws + EL * 12 + EL * 2 + (size_t)6 * NLc * Hc * Nc * 4);

  const int T = NLc * Hc * Nc;
  precompute_params<<<T / 256, 256, 0, stream>>>(logdt, Are, Aim, Cre, Cim, P);
  const int WN = NLc * 2 * Hc * Hc;
  to_bf16<<<WN / 256, 256, 0, stream>>>(outw, wb, WN);
  encode<<<(int)(EL / 256), 256, 0, stream>>>(x, enc_w, enc_b, hbuf);

  for (int i = 0; i < NLc; ++i) {
    layernorm<<<Mc, 256, 0, stream>>>(hbuf, lnw + i * Hc, lnb + i * Hc, bufA);
    transpose_f32<<<dim3(Hc / 32, Lc / 32, Bc), dim3(32, 8), 0, stream>>>(bufA, bufB);
    s4_scan<<<(Bc * Hc) / 4, 64, (size_t)8 * Lc * sizeof(float), stream>>>(
        bufB,
        P + 0 * T + i * Hc * Nc, P + 1 * T + i * Hc * Nc,
        P + 2 * T + i * Hc * Nc, P + 3 * T + i * Hc * Nc,
        P + 4 * T + i * Hc * Nc, P + 5 * T + i * Hc * Nc,
        Dp + i * Hc, (__bf16*)bufA);
    transpose_bf16<<<dim3(Lc / 32, Hc / 32, Bc), dim3(32, 8), 0, stream>>>(
        (const __bf16*)bufA, bufC);
    gemm_glu<<<dim3(Mc / 128, Hc / 16), 256, 0, stream>>>(
        bufC, wb + (size_t)i * 2 * Hc * Hc, outb + i * 2 * Hc, hbuf);
  }
  decode<<<Mc / 8, 256, 0, stream>>>(hbuf, decw, decb, out);
}
